// TRGTInternalRiskEncoder_79860621902504
// MI455X (gfx1250) — matrix operand order repaired and checked by reading, not compiled, not run on hardware
//
#include <hip/hip_runtime.h>
#include <hip/hip_bf16.h>
#include <math.h>

#define T_ 4096
#define H_ 128
#define N_ 262144
#define E_ 1048576
#define NET_ 8
static constexpr size_t TH = (size_t)T_ * H_;

typedef __attribute__((ext_vector_type(16))) __bf16 v16bf;
typedef __attribute__((ext_vector_type(8)))  float  v8f;

__device__ __forceinline__ float geluf(float x) {
  // exact GELU: 0.5*x*(1+erf(x/sqrt(2)))
  return 0.5f * x * (1.0f + erff(x * 0.70710678118654752f));
}

__device__ __forceinline__ float blockReduceSum(float v, float* sm) {
  const int tid = threadIdx.x;
  sm[tid] = v;
  __syncthreads();
  #pragma unroll
  for (int s = 64; s > 0; s >>= 1) {
    if (tid < s) sm[tid] += sm[tid + s];
    __syncthreads();
  }
  float r = sm[0];
  __syncthreads();
  return r;
}

// ---------------------------------------------------------------- zero init
__global__ void zero_kernel(float* __restrict__ ws, float* __restrict__ outn, int n) {
  int i = blockIdx.x * blockDim.x + threadIdx.x;
  const int stride = gridDim.x * blockDim.x;
  for (; i < n; i += stride) ws[i] = 0.0f;
  if (blockIdx.x == 0 && threadIdx.x < 3) outn[threadIdx.x] = 0.0f;
}

// --------------------------------------------- W2/W3 transpose + bf16 convert
__global__ void wprep_kernel(const float* __restrict__ W2, const float* __restrict__ W3,
                             __bf16* __restrict__ W2T, __bf16* __restrict__ W3T) {
  const int i = blockIdx.x * blockDim.x + threadIdx.x;
  const int stride = gridDim.x * blockDim.x;
  for (int idx = i; idx < 1056 * 256; idx += stride) {
    const int k = idx >> 8, n = idx & 255;            // W2 is [1056,256] row-major
    W2T[(size_t)n * 1056 + k] = (__bf16)W2[idx];
  }
  for (int idx = i; idx < 256 * 128; idx += stride) {
    const int k = idx >> 7, n = idx & 127;            // W3 is [256,128] row-major
    W3T[(size_t)n * 256 + k] = (__bf16)W3[idx];
  }
}

// ---------------------------------------------------------------- edge phase
// acc slots: 0 in_short, 1 out_short, 2 in_long, 3 out_long, 4 src_short, 5 src_long,
//            6 hop1, 7 hop2
// scal slots (each [T]): 0 ds_in, 1 ds_out, 2 dl_in, 3 dl_out, 4 tcount, 5 tsum,
//                        6 in_count, 7 out_count, 8 hop1_count, 9 hop2_count
__global__ void edge_accum_kernel(const float* __restrict__ node_repr,
                                  const float* __restrict__ edge_repr,
                                  const int* __restrict__ edge_src,
                                  const int* __restrict__ edge_dst,
                                  const int* __restrict__ rel_ids,
                                  const float* __restrict__ etime,
                                  const int* __restrict__ esg,
                                  float* __restrict__ acc,
                                  float* __restrict__ scal) {
  const int lane = threadIdx.x & 31;
  const int wid  = (blockIdx.x * blockDim.x + threadIdx.x) >> 5;
  const int nw   = (gridDim.x * blockDim.x) >> 5;
  for (int e = wid; e < E_; e += nw) {
    const int sg = esg[e];
    if (edge_dst[e] != sg) continue;      // target_local_idx[sg] == sg; non-target edges: no work
    const float t   = etime[e];
    const int   dir = (rel_ids[e] < NET_) ? 0 : 1;
    const float wS  = __expf(-t);
    const float wL  = __expf(-t * (1.0f / 24.0f));
    const int   src = edge_src[e];
    const int   h   = lane * 4;
    const float4 er = *(const float4*)(edge_repr + (size_t)e * H_ + h);
    const float4 nr = *(const float4*)(node_repr + (size_t)src * H_ + h);
    float* as_ = acc + (size_t)dir       * TH + (size_t)sg * H_ + h;
    float* al_ = acc + (size_t)(2 + dir) * TH + (size_t)sg * H_ + h;
    float* ss_ = acc + (size_t)4         * TH + (size_t)sg * H_ + h;
    float* sl_ = acc + (size_t)5         * TH + (size_t)sg * H_ + h;
    atomicAdd(as_ + 0, wS * er.x); atomicAdd(as_ + 1, wS * er.y);
    atomicAdd(as_ + 2, wS * er.z); atomicAdd(as_ + 3, wS * er.w);
    atomicAdd(al_ + 0, wL * er.x); atomicAdd(al_ + 1, wL * er.y);
    atomicAdd(al_ + 2, wL * er.z); atomicAdd(al_ + 3, wL * er.w);
    atomicAdd(ss_ + 0, wS * nr.x); atomicAdd(ss_ + 1, wS * nr.y);
    atomicAdd(ss_ + 2, wS * nr.z); atomicAdd(ss_ + 3, wS * nr.w);
    atomicAdd(sl_ + 0, wL * nr.x); atomicAdd(sl_ + 1, wL * nr.y);
    atomicAdd(sl_ + 2, wL * nr.z); atomicAdd(sl_ + 3, wL * nr.w);
    if (lane == 0) {
      atomicAdd(scal + dir * T_ + sg, wS);
      atomicAdd(scal + (2 + dir) * T_ + sg, wL);
      atomicAdd(scal + 4 * T_ + sg, 1.0f);
      atomicAdd(scal + 5 * T_ + sg, t);
      atomicAdd(scal + (6 + dir) * T_ + sg, 1.0f);
    }
  }
}

// ---------------------------------------------------------------- node phase
__global__ void node_accum_kernel(const float* __restrict__ node_repr,
                                  const int* __restrict__ nsg,
                                  const int* __restrict__ hop,
                                  float* __restrict__ acc,
                                  float* __restrict__ scal) {
  const int lane = threadIdx.x & 31;
  const int wid  = (blockIdx.x * blockDim.x + threadIdx.x) >> 5;
  const int nw   = (gridDim.x * blockDim.x) >> 5;
  for (int nno = wid; nno < N_; nno += nw) {
    const int sg   = nsg[nno];
    const int slot = (hop[nno] == 1) ? 6 : 7;   // depth in {1,2,3}: exactly one bucket
    const int h    = lane * 4;
    const float4 v = *(const float4*)(node_repr + (size_t)nno * H_ + h);
    float* p = acc + (size_t)slot * TH + (size_t)sg * H_ + h;
    atomicAdd(p + 0, v.x); atomicAdd(p + 1, v.y);
    atomicAdd(p + 2, v.z); atomicAdd(p + 3, v.w);
    if (lane == 0) atomicAdd(scal + (slot - 6 + 8) * T_ + sg, 1.0f);
  }
}

// ----------------------------------------------- per-target features + norms
__global__ void feat_kernel(const float* __restrict__ acc, const float* __restrict__ scal,
                            const float* __restrict__ W1, const float* __restrict__ b1,
                            __bf16* __restrict__ fused, float* __restrict__ out_norms) {
  __shared__ float sm[128];
  const int t = blockIdx.x, h = threadIdx.x;
  const float ds_in = scal[0 * T_ + t], ds_out = scal[1 * T_ + t];
  const float dl_in = scal[2 * T_ + t], dl_out = scal[3 * T_ + t];
  const float tcnt  = scal[4 * T_ + t], tsum   = scal[5 * T_ + t];
  const float icnt  = scal[6 * T_ + t], ocnt   = scal[7 * T_ + t];
  const float h1c   = scal[8 * T_ + t], h2c    = scal[9 * T_ + t];
  const size_t base = (size_t)t * H_ + h;

  const float in_s  = acc[0 * TH + base] / fmaxf(ds_in, 1e-6f);
  const float out_s = acc[1 * TH + base] / fmaxf(ds_out, 1e-6f);
  const float in_l  = acc[2 * TH + base] / fmaxf(dl_in, 1e-6f);
  const float out_l = acc[3 * TH + base] / fmaxf(dl_out, 1e-6f);
  const float smass = ds_in + ds_out, lmass = dl_in + dl_out;
  const float src_s = acc[4 * TH + base] / fmaxf(smass, 1e-6f);
  const float src_l = acc[5 * TH + base] / fmaxf(lmass, 1e-6f);
  const float h1    = acc[6 * TH + base] / fmaxf(h1c, 1e-6f);
  const float h2    = acc[7 * TH + base] / fmaxf(h2c, 1e-6f);

  float xs[8];
  xs[0] = in_s + out_s - in_l - out_l;  // burst_delta
  xs[1] = out_l - in_l;                 // direction_gap
  xs[2] = h1 - h2;                      // hop_gap
  xs[3] = src_s - src_l;                // short_long_gap
  xs[5] = h1;                           // hop1_mean
  xs[6] = h2;                           // hop2_mean
  xs[7] = in_l + out_l;                 // risk_base

  __bf16* frow = fused + (size_t)t * 1056;

  // direction_gap first: asymmetry = LN(|LN(direction_gap)|)
  {
    const float x = xs[1];
    const float s = blockReduceSum(x, sm);
    const float q = blockReduceSum(x * x, sm);
    const float m = s * (1.0f / H_);
    const float v = fmaxf(q * (1.0f / H_) - m * m, 0.0f);
    const float y = (x - m) * rsqrtf(v + 1e-5f);
    frow[1 * H_ + h] = (__bf16)y;
    xs[4] = fabsf(y);
    if (h == 0) atomicAdd(out_norms + 2, sqrtf((float)H_ * v / (v + 1e-5f)) * (1.0f / T_));
  }
  for (int f = 0; f < 8; ++f) {
    if (f == 1) continue;
    const float x = xs[f];
    const float s = blockReduceSum(x, sm);
    const float q = blockReduceSum(x * x, sm);
    const float m = s * (1.0f / H_);
    const float v = fmaxf(q * (1.0f / H_) - m * m, 0.0f);
    const float y = (x - m) * rsqrtf(v + 1e-5f);
    frow[f * H_ + h] = (__bf16)y;
    if (h == 0) {
      if (f == 2) atomicAdd(out_norms + 0, sqrtf((float)H_ * v / (v + 1e-5f)) * (1.0f / T_));
      if (f == 3) atomicAdd(out_norms + 1, sqrtf((float)H_ * v / (v + 1e-5f)) * (1.0f / T_));
    }
  }
  // scalar embedding: gelu(sf @ W1 + b1), 32 outputs
  if (h < 32) {
    const float sf[8] = { log1pf(icnt), log1pf(ocnt), log1pf(h1c), log1pf(h2c),
                          tsum / fmaxf(tcnt, 1e-6f), smass, lmass, smass - lmass };
    float a = b1[h];
    #pragma unroll
    for (int i = 0; i < 8; ++i) a += sf[i] * W1[i * 32 + h];
    frow[1024 + h] = (__bf16)geluf(a);
  }
}

// -------------------------------------------------- GEMM1: [4096,1056]x[1056,256]
__global__ void gemm1_kernel(const __bf16* __restrict__ A,   // [T_,1056] fused (bf16, row-major)
                             const __bf16* __restrict__ Bt,  // [256,1056] = W2^T (bf16)
                             const float* __restrict__ bias, // [256]
                             __bf16* __restrict__ Hid) {     // [T_,256]
  const int lane = threadIdx.x & 31;
  const int wave = blockIdx.x * (blockDim.x >> 5) + (threadIdx.x >> 5);
  const int NT = 16;                       // 256/16 N-tiles
  const int mt = wave / NT, nt = wave % NT;
  const int half = lane >> 4;              // K-half select per ISA 16-bit fragment layout
  const __bf16* arow = A  + (size_t)(mt * 16 + (lane & 15)) * 1056 + half * 8;
  const __bf16* brow = Bt + (size_t)(nt * 16 + (lane & 15)) * 1056 + half * 8;
  v8f c = {};
  for (int kt = 0; kt < 33; ++kt) {        // K = 1056 = 33*32
    v16bf a, b;
    const __bf16* ap = arow + kt * 32;
    const __bf16* bp = brow + kt * 32;
    __builtin_prefetch(ap + 32, 0, 0);
    __builtin_prefetch(bp + 32, 0, 0);
    #pragma unroll
    for (int i = 0; i < 8; ++i) { a[i] = ap[i]; a[8 + i] = ap[16 + i]; }
    #pragma unroll
    for (int i = 0; i < 8; ++i) { b[i] = bp[i]; b[8 + i] = bp[16 + i]; }
    c = __builtin_amdgcn_wmma_f32_16x16x32_bf16(false, a, false, b, (short)0, c, false, false);
  }
  const int n = nt * 16 + (lane & 15);
  const float bn = bias[n];
  #pragma unroll
  for (int j = 0; j < 8; ++j) {
    const int m = mt * 16 + j + half * 8;
    Hid[(size_t)m * 256 + n] = (__bf16)geluf(c[j] + bn);
  }
}

// -------------------------------------------------- GEMM2: [4096,256]x[256,128]
__global__ void gemm2_kernel(const __bf16* __restrict__ A,   // [T_,256] hidden (bf16)
                             const __bf16* __restrict__ Bt,  // [128,256] = W3^T (bf16)
                             const float* __restrict__ bias, // [128]
                             float* __restrict__ out) {      // [T_,128] f32
  const int lane = threadIdx.x & 31;
  const int wave = blockIdx.x * (blockDim.x >> 5) + (threadIdx.x >> 5);
  const int NT = 8;                        // 128/16 N-tiles
  const int mt = wave / NT, nt = wave % NT;
  const int half = lane >> 4;
  const __bf16* arow = A  + (size_t)(mt * 16 + (lane & 15)) * 256 + half * 8;
  const __bf16* brow = Bt + (size_t)(nt * 16 + (lane & 15)) * 256 + half * 8;
  v8f c = {};
  #pragma unroll
  for (int kt = 0; kt < 8; ++kt) {         // K = 256 = 8*32
    v16bf a, b;
    const __bf16* ap = arow + kt * 32;
    const __bf16* bp = brow + kt * 32;
    #pragma unroll
    for (int i = 0; i < 8; ++i) { a[i] = ap[i]; a[8 + i] = ap[16 + i]; }
    #pragma unroll
    for (int i = 0; i < 8; ++i) { b[i] = bp[i]; b[8 + i] = bp[16 + i]; }
    c = __builtin_amdgcn_wmma_f32_16x16x32_bf16(false, a, false, b, (short)0, c, false, false);
  }
  const int n = nt * 16 + (lane & 15);
  const float bn = bias[n];
  #pragma unroll
  for (int j = 0; j < 8; ++j) {
    const int m = mt * 16 + j + half * 8;
    out[(size_t)m * H_ + n] = 0.25f * tanhf(c[j] + bn);
  }
}

// ---------------------------------------------------------------------------
extern "C" void kernel_launch(void* const* d_in, const int* in_sizes, int n_in,
                              void* d_out, int out_size, void* d_ws, size_t ws_size,
                              hipStream_t stream) {
  const float* node_repr = (const float*)d_in[0];
  const float* edge_repr = (const float*)d_in[1];
  const int*   edge_src  = (const int*)d_in[2];
  const int*   edge_dst  = (const int*)d_in[3];
  const int*   rel_ids   = (const int*)d_in[4];
  const float* etime     = (const float*)d_in[5];
  // d_in[6] = target_local_idx == arange(T): identity, unused
  const int*   nsg       = (const int*)d_in[7];
  const int*   esg       = (const int*)d_in[8];
  const int*   hop       = (const int*)d_in[9];
  const float* W1        = (const float*)d_in[10];
  const float* b1        = (const float*)d_in[11];
  const float* W2        = (const float*)d_in[12];
  const float* b2        = (const float*)d_in[13];
  const float* W3        = (const float*)d_in[14];
  const float* b3        = (const float*)d_in[15];
  float* out = (float*)d_out;                 // [T*H] risk + 3 norms

  // workspace layout (float units)
  float* ws   = (float*)d_ws;
  float* acc  = ws;                           // 8 * T * H
  float* scal = ws + 8 * TH;                  // 10 * T
  size_t off = 8 * TH + 10 * (size_t)T_;
  off = (off + 7) & ~(size_t)7;               // 32B align for bf16 vector loads
  __bf16* fused = (__bf16*)(ws + off);        // T * 1056 bf16
  off += ((size_t)T_ * 1056) / 2;
  __bf16* hid   = (__bf16*)(ws + off);        // T * 256 bf16
  off += ((size_t)T_ * 256) / 2;
  __bf16* W2T   = (__bf16*)(ws + off);        // 256 * 1056 bf16
  off += (size_t)(256 * 1056) / 2;
  __bf16* W3T   = (__bf16*)(ws + off);        // 128 * 256 bf16

  const int zero_n = (int)(8 * TH + 10 * (size_t)T_);
  zero_kernel<<<4096, 256, 0, stream>>>(ws, out + TH, zero_n);
  wprep_kernel<<<512, 256, 0, stream>>>(W2, W3, W2T, W3T);
  edge_accum_kernel<<<4096, 256, 0, stream>>>(node_repr, edge_repr, edge_src, edge_dst,
                                              rel_ids, etime, esg, acc, scal);
  node_accum_kernel<<<2048, 256, 0, stream>>>(node_repr, nsg, hop, acc, scal);
  feat_kernel<<<T_, 128, 0, stream>>>(acc, scal, W1, b1, fused, out + TH);
  gemm1_kernel<<<512, 256, 0, stream>>>(fused, W2T, b2, hid);   // 256*16 tiles / 8 waves
  gemm2_kernel<<<256, 256, 0, stream>>>(hid, W3T, b3, out);     // 256*8 tiles / 8 waves
}
